// IPEXGPTJAttention_80616536146209
// MI455X (gfx1250) — matrix operand order repaired and checked by reading, not compiled, not run on hardware
//
#include <hip/hip_runtime.h>
#include <hip/hip_bf16.h>
#include <math.h>

// ---------------------------------------------------------------------------
// GPT-J attention for MI455X (gfx1250, wave32, WMMA).
// All matrix math runs through v_wmma_f32_16x16x32_bf16 (bf16 in, f32 acc).
// Staging uses CDNA5 async memory->LDS DMA (global_load_async_to_lds_b128,
// ASYNCcnt) with double-buffered LDS in both the GEMM and flash attention.
// ---------------------------------------------------------------------------

typedef __bf16 bf16;
typedef __attribute__((ext_vector_type(16))) __bf16 v16bf;
typedef __attribute__((ext_vector_type(8)))  __bf16 v8bf;
typedef __attribute__((ext_vector_type(8)))  float  v8f;

#define B_  2
#define S_  2048
#define D_  4096
#define H_  16
#define HD_ 256
#define M_  (B_*S_)   // 4096 rows of X

__device__ __forceinline__ bf16 f2bf(float f) {
  unsigned u = __builtin_bit_cast(unsigned, f);
  unsigned r = u + 0x7FFFu + ((u >> 16) & 1u);   // round-to-nearest-even
  unsigned short hs = (unsigned short)(r >> 16);
  return __builtin_bit_cast(bf16, hs);
}
__device__ __forceinline__ float bf2f(bf16 x) {
  unsigned short hs = __builtin_bit_cast(unsigned short, x);
  return __builtin_bit_cast(float, (unsigned)hs << 16);
}

// CDNA5 async DMA: memory -> LDS, 16B per lane, tracked by ASYNCcnt.
// VDST = VGPR holding LDS byte offset (generic addr low 32 bits == LDS offset),
// VADDR = 64-bit global address, SADDR = off (GV mode).
__device__ __forceinline__ void async_ld_b128(void* lds, const void* gptr) {
  unsigned lds_off = (unsigned)(size_t)lds;
  asm volatile("global_load_async_to_lds_b128 %0, %1, off"
               :: "v"(lds_off), "v"(gptr) : "memory");
}
__device__ __forceinline__ void wait_async0() {
  asm volatile("s_wait_asynccnt 0x0" ::: "memory");
}

// ---------------------------------------------------------------------------
// 1) f32 -> bf16 bulk convert (hidden states)
// ---------------------------------------------------------------------------
__global__ __launch_bounds__(256) void cvt_f32_bf16(const float* __restrict__ src,
                                                    bf16* __restrict__ dst, int n) {
  int i = (blockIdx.x * 256 + threadIdx.x) * 4;
  if (i + 3 < n) {
    float4 v = *(const float4*)(src + i);
    dst[i + 0] = f2bf(v.x);
    dst[i + 1] = f2bf(v.y);
    dst[i + 2] = f2bf(v.z);
    dst[i + 3] = f2bf(v.w);
  }
}

// ---------------------------------------------------------------------------
// 2) W [K][N] f32 -> W^T [N][K] bf16 (so WMMA B-fragments read contiguous K)
// ---------------------------------------------------------------------------
__global__ __launch_bounds__(256) void transpose_cvt(const float* __restrict__ W,
                                                     bf16* __restrict__ WT) {
  __shared__ float t[32][33];
  const int tx = threadIdx.x, ty = threadIdx.y;
  const int bx = blockIdx.x * 32;  // n
  const int by = blockIdx.y * 32;  // k
#pragma unroll
  for (int j = 0; j < 4; ++j)
    t[ty + j*8][tx] = W[(size_t)(by + ty + j*8) * D_ + bx + tx];
  __syncthreads();
#pragma unroll
  for (int j = 0; j < 4; ++j)
    WT[(size_t)(bx + ty + j*8) * D_ + by + tx] = f2bf(t[tx][ty + j*8]);
}

// ---------------------------------------------------------------------------
// 3) WMMA GEMM: C = A[M][K] * B (B given transposed as BT[N][K]), bf16->f32.
//    Block tile 128x128, BK=32, 8 waves (4x2), each wave 2x4 16x16 tiles.
//    Double-buffered LDS, async-DMA staging, one barrier per K-step.
//    mode 0: bf16 out, [b][h][s][hd]   (Q, K layout for QK^T)
//    mode 1: bf16 out, [b][h][hd][s]   (V^T layout for PV)
//    mode 2: f32 out,  row-major [M][N] (final output)
// ---------------------------------------------------------------------------
#define TM 128
#define TN 128
#define TK 32
#define LDA_ 40   // LDS row stride (elements): 80B, 16B-aligned, conflict-free

__global__ __launch_bounds__(256) void gemm_wmma_bf16(
    const bf16* __restrict__ A, const bf16* __restrict__ BT,
    void* __restrict__ Cout, int K, int mode)
{
  __shared__ bf16 As[2][TM * LDA_];
  __shared__ bf16 Bs[2][TN * LDA_];
  const int tid  = threadIdx.x;
  const int lane = tid & 31, wid = tid >> 5;
  const int hl   = lane >> 4, l16 = lane & 15;
  const int wm   = wid >> 1,  wn  = wid & 1;
  const int bm   = blockIdx.y * TM, bn = blockIdx.x * TN;

  const v8f vz = {0,0,0,0,0,0,0,0};
  v8f acc[2][4];
#pragma unroll
  for (int mt = 0; mt < 2; ++mt)
#pragma unroll
    for (int nt = 0; nt < 4; ++nt) acc[mt][nt] = vz;

  // per-thread staging coordinates (512 16B-chunks per 128x32 tile)
  const int r0 = tid >> 2,          c0 = (tid & 3) << 3;
  const int r1 = (tid + 256) >> 2,  c1 = ((tid + 256) & 3) << 3;

  const int nIter = K / TK;
  // prologue: DMA tile 0 into buffer 0
  {
    async_ld_b128(&As[0][r0*LDA_ + c0], A  + (size_t)(bm+r0)*K + c0);
    async_ld_b128(&Bs[0][r0*LDA_ + c0], BT + (size_t)(bn+r0)*K + c0);
    async_ld_b128(&As[0][r1*LDA_ + c1], A  + (size_t)(bm+r1)*K + c1);
    async_ld_b128(&Bs[0][r1*LDA_ + c1], BT + (size_t)(bn+r1)*K + c1);
  }

  for (int it = 0; it < nIter; ++it) {
    const int cur = it & 1;
    wait_async0();        // this wave's DMA for buffer `cur` has landed
    __syncthreads();      // all waves' DMA landed; all waves done reading buf cur^1
    if (it + 1 < nIter) { // overwrite the buffer everyone just finished reading
      const int nk = (it + 1) * TK;
      async_ld_b128(&As[cur^1][r0*LDA_ + c0], A  + (size_t)(bm+r0)*K + nk + c0);
      async_ld_b128(&Bs[cur^1][r0*LDA_ + c0], BT + (size_t)(bn+r0)*K + nk + c0);
      async_ld_b128(&As[cur^1][r1*LDA_ + c1], A  + (size_t)(bm+r1)*K + nk + c1);
      async_ld_b128(&Bs[cur^1][r1*LDA_ + c1], BT + (size_t)(bn+r1)*K + nk + c1);
    }

    // A-fragment: M = l16, K chunks at elem 8*hl and 16+8*hl
    v16bf af[2];
#pragma unroll
    for (int mt = 0; mt < 2; ++mt) {
      const bf16* p = &As[cur][(wm*32 + mt*16 + l16) * LDA_ + 8*hl];
      v8bf lo = *(const v8bf*)p;
      v8bf hi = *(const v8bf*)(p + 16);
#pragma unroll
      for (int e = 0; e < 8; ++e) { af[mt][e] = lo[e]; af[mt][e+8] = hi[e]; }
    }
    // B-fragment: N = l16, K chunks at elem 8*hl and 16+8*hl
    v16bf bfm[4];
#pragma unroll
    for (int nt = 0; nt < 4; ++nt) {
      const bf16* p = &Bs[cur][(wn*64 + nt*16 + l16) * LDA_ + 8*hl];
      v8bf lo = *(const v8bf*)p;
      v8bf hi = *(const v8bf*)(p + 16);
#pragma unroll
      for (int e = 0; e < 8; ++e) { bfm[nt][e] = lo[e]; bfm[nt][e+8] = hi[e]; }
    }
#pragma unroll
    for (int mt = 0; mt < 2; ++mt)
#pragma unroll
      for (int nt = 0; nt < 4; ++nt)
        acc[mt][nt] = __builtin_amdgcn_wmma_f32_16x16x32_bf16(
            false, af[mt], false, bfm[nt], (short)0, acc[mt][nt], false, false);
  }

  // D layout: col = l16 (+tile), row = r + 8*hl (+tile)
#pragma unroll
  for (int mt = 0; mt < 2; ++mt)
#pragma unroll
    for (int nt = 0; nt < 4; ++nt)
#pragma unroll
      for (int r = 0; r < 8; ++r) {
        int grow = bm + wm*32 + mt*16 + r + 8*hl;
        int gcol = bn + wn*64 + nt*16 + l16;
        float v = acc[mt][nt][r];
        if (mode == 2) {
          ((float*)Cout)[(size_t)grow * D_ + gcol] = v;
        } else {
          int b  = grow >> 11, s  = grow & (S_-1);
          int h  = gcol >> 8,  hd = gcol & (HD_-1);
          size_t idx = (mode == 0)
            ? ((((size_t)(b*H_ + h))*S_  + s )*HD_ + hd)
            : ((((size_t)(b*H_ + h))*HD_ + hd)*S_  + s );
          ((bf16*)Cout)[idx] = f2bf(v);
        }
      }
}

// ---------------------------------------------------------------------------
// 4) RoPE on first ROT=64 dims of Q,K (layout [b][h][s][hd], bf16 in place)
// ---------------------------------------------------------------------------
__global__ __launch_bounds__(256) void rope_kernel(
    bf16* __restrict__ Q, bf16* __restrict__ Kt,
    const float* __restrict__ emb, const int* __restrict__ pos_ids)
{
  int idx = blockIdx.x * 256 + threadIdx.x;    // B*H*S*32 pair-lanes
  int i = idx & 31;
  int s = (idx >> 5)  & (S_-1);
  int h = (idx >> 16) & (H_-1);
  int b = idx >> 20;
  if (b >= B_) return;
  int pos  = pos_ids[b*S_ + s];
  float sn = emb[pos*64 + i];
  float cs = emb[pos*64 + 32 + i];
  size_t base = (((size_t)(b*H_ + h))*S_ + s)*HD_ + 2*i;
  float x1 = bf2f(Q[base]), x2 = bf2f(Q[base+1]);
  Q[base]     = f2bf(x1*cs - x2*sn);
  Q[base + 1] = f2bf(x2*cs + x1*sn);
  float y1 = bf2f(Kt[base]), y2 = bf2f(Kt[base+1]);
  Kt[base]     = f2bf(y1*cs - y2*sn);
  Kt[base + 1] = f2bf(y2*cs + y1*sn);
}

// ---------------------------------------------------------------------------
// 5) Flash attention. Block = (128 q-rows, one (b,h)); 8 waves x 16 q-rows.
//    Q fragments resident in VGPRs; K (32x256) and V^T (256x32) async-DMA'd
//    into double-buffered LDS; one barrier per key tile.
//    Online softmax; probs reshaped C-layout -> A-layout via wave-private LDS.
// ---------------------------------------------------------------------------
#define AQ 128
#define AK 32
#define KSTR 264   // 528B rows: 16B aligned, conflict-free
#define VSTR 40
#define PSTR 40

__global__ __launch_bounds__(256) void flash_attn(
    const bf16* __restrict__ Q, const bf16* __restrict__ Kc,
    const bf16* __restrict__ VT, const float* __restrict__ amask,
    bf16* __restrict__ Ao)
{
  __shared__ bf16 Ks[2][AK  * KSTR];   // 2 x 16.9 KB
  __shared__ bf16 Vs[2][HD_ * VSTR];   // 2 x 20.5 KB
  __shared__ bf16 Ps[8 * 16 * PSTR];   // 10.2 KB (per-wave 16x32 scratch)

  const int tid  = threadIdx.x;
  const int lane = tid & 31, wid = tid >> 5;
  const int hl   = lane >> 4, l16 = lane & 15;
  const int bh   = blockIdx.y;
  const int b    = bh >> 4, h = bh & 15;
  const int qbase = blockIdx.x * AQ;

  const bf16* Qg = Q  + ((size_t)bh * S_ + qbase) * HD_;
  const bf16* Kg = Kc + (size_t)bh * S_ * HD_;
  const bf16* Vg = VT + (size_t)bh * HD_ * S_;

  // async-DMA one 32-key tile (K rows + V^T columns) into LDS buffer `buf`
  auto issue_tile = [&](int buf, int skb) {
#pragma unroll
    for (int i = 0; i < 4; ++i) {
      int c  = tid + i * 256;
      int kr = c >> 5, kc = (c & 31) << 3;
      async_ld_b128(&Ks[buf][kr*KSTR + kc], Kg + (size_t)(skb + kr)*HD_ + kc);
      int vr = c >> 2, vc = (c & 3) << 3;
      async_ld_b128(&Vs[buf][vr*VSTR + vc], Vg + (size_t)vr*S_ + skb + vc);
    }
  };

  // Resident Q A-fragments: 16 rows x 256 hd = 8 fragments
  v16bf qf[8];
  {
    const bf16* qrow = Qg + (size_t)(wid*16 + l16) * HD_;
#pragma unroll
    for (int kk = 0; kk < 8; ++kk) {
      v8bf lo = *(const v8bf*)(qrow + kk*32 + 8*hl);
      v8bf hi = *(const v8bf*)(qrow + kk*32 + 16 + 8*hl);
#pragma unroll
      for (int e = 0; e < 8; ++e) { qf[kk][e] = lo[e]; qf[kk][e+8] = hi[e]; }
    }
  }

  const v8f vz = {0,0,0,0,0,0,0,0};
  float m_i[8], l_i[8];
  v8f o[16];
#pragma unroll
  for (int r = 0; r < 8; ++r) { m_i[r] = -INFINITY; l_i[r] = 0.0f; }
#pragma unroll
  for (int t = 0; t < 16; ++t) o[t] = vz;

  const int qrow_max = qbase + wid*16 + 15;
  const int nTiles = (qbase + AQ) / AK;   // causal upper bound for this block

  issue_tile(0, 0);   // prologue DMA

  for (int it = 0; it < nTiles; ++it) {
    const int skb = it * AK;
    const int cur = it & 1;
    wait_async0();        // this wave's DMA for buffer `cur` has landed
    __syncthreads();      // all waves landed; all waves done reading buf cur^1
    if (it + 1 < nTiles)  // overwrite the buffer everyone just finished reading
      issue_tile(cur ^ 1, skb + AK);

    if (skb <= qrow_max) {    // wave-uniform causal skip
      // scores: S = Q K^T  (2 tiles of 16 keys)
      v8f st[2] = {vz, vz};
#pragma unroll
      for (int kk = 0; kk < 8; ++kk) {
#pragma unroll
        for (int nt = 0; nt < 2; ++nt) {
          const bf16* p = &Ks[cur][(nt*16 + l16)*KSTR + kk*32 + 8*hl];
          v8bf lo = *(const v8bf*)p;
          v8bf hi = *(const v8bf*)(p + 16);
          v16bf kf;
#pragma unroll
          for (int e = 0; e < 8; ++e) { kf[e] = lo[e]; kf[e+8] = hi[e]; }
          st[nt] = __builtin_amdgcn_wmma_f32_16x16x32_bf16(
              false, qf[kk], false, kf, (short)0, st[nt], false, false);
        }
      }
      // mask + online softmax (row lives in 16 lanes of this half-wave group)
#pragma unroll
      for (int r = 0; r < 8; ++r) {
        int m = r + 8*hl;
        int grow = qbase + wid*16 + m;
        int g0 = skb + l16, g1 = skb + 16 + l16;
        float s0 = (g0 <= grow) ? st[0][r] * 0.0625f : -4094.0f;  // MASK/sqrt(HD)
        float s1 = (g1 <= grow) ? st[1][r] * 0.0625f : -4094.0f;
        s0 += amask[b*S_ + g0];
        s1 += amask[b*S_ + g1];
        float cm = fmaxf(s0, s1);
#pragma unroll
        for (int d = 1; d < 16; d <<= 1) cm = fmaxf(cm, __shfl_xor(cm, d, 32));
        float nm    = fmaxf(m_i[r], cm);
        float scale = __expf(m_i[r] - nm);
        m_i[r] = nm;
        float p0 = __expf(s0 - nm);
        float p1 = __expf(s1 - nm);
        float ps = p0 + p1;
#pragma unroll
        for (int d = 1; d < 16; d <<= 1) ps += __shfl_xor(ps, d, 32);
        l_i[r] = l_i[r] * scale + ps;
#pragma unroll
        for (int t = 0; t < 16; ++t) o[t][r] *= scale;
        bf16* prow = &Ps[(wid*16 + m) * PSTR];
        prow[l16]      = f2bf(p0);
        prow[16 + l16] = f2bf(p1);
      }
      asm volatile("s_wait_dscnt 0x0" ::: "memory");  // LDS reshape fence (same wave)

      // P A-fragment (16 q x 32 sk)
      v16bf pf;
      {
        const bf16* p = &Ps[(wid*16 + l16) * PSTR + 8*hl];
        v8bf lo = *(const v8bf*)p;
        v8bf hi = *(const v8bf*)(p + 16);
#pragma unroll
        for (int e = 0; e < 8; ++e) { pf[e] = lo[e]; pf[e+8] = hi[e]; }
      }
      // O += P @ V  (16 hd-tiles)
#pragma unroll
      for (int ht = 0; ht < 16; ++ht) {
        const bf16* p = &Vs[cur][(ht*16 + l16)*VSTR + 8*hl];
        v8bf lo = *(const v8bf*)p;
        v8bf hi = *(const v8bf*)(p + 16);
        v16bf vf;
#pragma unroll
        for (int e = 0; e < 8; ++e) { vf[e] = lo[e]; vf[e+8] = hi[e]; }
        o[ht] = __builtin_amdgcn_wmma_f32_16x16x32_bf16(
            false, pf, false, vf, (short)0, o[ht], false, false);
      }
    }
  }

  // epilogue: O / l_i -> attn output [b][s][h*256+hd] as bf16
#pragma unroll
  for (int r = 0; r < 8; ++r) {
    float inv = 1.0f / l_i[r];
    int s = qbase + wid*16 + r + 8*hl;
#pragma unroll
    for (int ht = 0; ht < 16; ++ht) {
      int col = h*HD_ + ht*16 + l16;
      Ao[((size_t)b*S_ + s)*D_ + col] = f2bf(o[ht][r] * inv);
    }
  }
}

// ---------------------------------------------------------------------------
// Launch
// ---------------------------------------------------------------------------
extern "C" void kernel_launch(void* const* d_in, const int* in_sizes, int n_in,
                              void* d_out, int out_size, void* d_ws, size_t ws_size,
                              hipStream_t stream)
{
  (void)in_sizes; (void)n_in; (void)out_size; (void)ws_size;
  const float* hidden = (const float*)d_in[0];
  const float* Wq  = (const float*)d_in[1];
  const float* Wk  = (const float*)d_in[2];
  const float* Wv  = (const float*)d_in[3];
  const float* Wo  = (const float*)d_in[4];
  const float* emb = (const float*)d_in[5];
  const float* amask = (const float*)d_in[6];
  const int*   pos = (const int*)d_in[7];

  char* ws = (char*)d_ws;
  const size_t SZ = (size_t)M_ * D_ * sizeof(bf16);   // 32 MB per buffer
  bf16* Xb  = (bf16*)(ws + 0*SZ);
  bf16* WqT = (bf16*)(ws + 1*SZ);
  bf16* WkT = (bf16*)(ws + 2*SZ);
  bf16* WvT = (bf16*)(ws + 3*SZ);
  bf16* WoT = (bf16*)(ws + 4*SZ);
  bf16* Qb  = (bf16*)(ws + 5*SZ);
  bf16* Kb  = (bf16*)(ws + 6*SZ);
  bf16* Vt  = (bf16*)(ws + 7*SZ);   // V^T [b][h][hd][s]
  bf16* At  = (bf16*)(ws + 8*SZ);   // attention out [b][s][d]

  // 1) precision conversion
  cvt_f32_bf16<<<(M_*D_)/1024, 256, 0, stream>>>(hidden, Xb, M_*D_);
  dim3 tb(32, 8), tg(D_/32, D_/32);
  transpose_cvt<<<tg, tb, 0, stream>>>(Wq, WqT);
  transpose_cvt<<<tg, tb, 0, stream>>>(Wk, WkT);
  transpose_cvt<<<tg, tb, 0, stream>>>(Wv, WvT);
  transpose_cvt<<<tg, tb, 0, stream>>>(Wo, WoT);

  // 2) QKV projections (WMMA)
  dim3 gg(D_/TN, M_/TM);
  gemm_wmma_bf16<<<gg, 256, 0, stream>>>(Xb, WqT, Qb, D_, 0);
  gemm_wmma_bf16<<<gg, 256, 0, stream>>>(Xb, WkT, Kb, D_, 0);
  gemm_wmma_bf16<<<gg, 256, 0, stream>>>(Xb, WvT, Vt, D_, 1);

  // 3) RoPE
  rope_kernel<<<(B_*H_*S_*32)/256, 256, 0, stream>>>(Qb, Kb, emb, pos);

  // 4) causal flash attention (WMMA, double-buffered async staging)
  dim3 ag(S_/AQ, B_*H_);
  flash_attn<<<ag, 256, 0, stream>>>(Qb, Kb, Vt, amask, At);

  // 5) output projection (WMMA, f32 out)
  gemm_wmma_bf16<<<gg, 256, 0, stream>>>(At, WoT, d_out, D_, 2);
}
